// GAT_19645180412673
// MI455X (gfx1250) — matrix operand order repaired and checked by reading, not compiled, not run on hardware
//
#include <hip/hip_runtime.h>
#include <hip/hip_bf16.h>
#include <math.h>

// ---------------- problem constants (from reference) ----------------
#define N_NODES 50000
#define N_EDGES 800000
#define E_TOT   (N_EDGES + N_NODES)   // self-loops appended
#define IN_CH   128
#define F1      128                   // HEADS*HID
#define HEADS   8
#define HID     16
#define OUT_CH  2
#define NEG_SLOPE 0.2f

typedef __bf16 bf16_t;
typedef __bf16 bf8v  __attribute__((ext_vector_type(8)));
typedef __bf16 bf16v __attribute__((ext_vector_type(16)));
typedef float  f8v   __attribute__((ext_vector_type(8)));

// order-preserving float <-> u32 key (for atomic segment-max on floats)
static __device__ __forceinline__ unsigned fkey(float f) {
  unsigned b = __float_as_uint(f);
  return b ^ ((b >> 31) ? 0xFFFFFFFFu : 0x80000000u);
}
static __device__ __forceinline__ float fkeyinv(unsigned k) {
  unsigned b = k ^ ((k >> 31) ? 0x80000000u : 0xFFFFFFFFu);
  return __uint_as_float(b);
}
#define NEG_INF_KEY 0x007FFFFFu   // fkey(-inf)

static __device__ __forceinline__ void atomAddF(float* p, float v) {
  __hip_atomic_fetch_add(p, v, __ATOMIC_RELAXED, __HIP_MEMORY_SCOPE_AGENT);
}
static __device__ __forceinline__ void atomMaxU(unsigned* p, unsigned v) {
  __hip_atomic_fetch_max(p, v, __ATOMIC_RELAXED, __HIP_MEMORY_SCOPE_AGENT);
}

// ---------------- prep kernels ----------------
__global__ void k_f32_to_bf16(const float* __restrict__ s, bf16_t* __restrict__ d, int n) {
  int i = blockIdx.x * blockDim.x + threadIdx.x;
  if (i < n) d[i] = (bf16_t)s[i];
}

// W1 [K=128, N=128] row-major f32  ->  Wt [N][K] col-major bf16 (so B fragments are contiguous in K)
__global__ void k_w1_bf16_t(const float* __restrict__ W, bf16_t* __restrict__ Wt) {
  int i = blockIdx.x * blockDim.x + threadIdx.x;
  if (i < IN_CH * F1) {
    int k = i / F1, n = i - k * F1;
    Wt[n * IN_CH + k] = (bf16_t)W[i];
  }
}

// ---------------- GEMM1: h = x @ W1 via v_wmma_f32_16x16x32_bf16 ----------------
// One wave computes one 16x16 tile of C. A: [M,128] bf16 row-major. Bt: [128 cols][128 K] bf16.
__global__ void k_gemm1_wmma(const bf16_t* __restrict__ A, const bf16_t* __restrict__ Bt,
                             float* __restrict__ C) {
  int gtid = blockIdx.x * blockDim.x + threadIdx.x;
  int wave = gtid >> 5;
  int lane = gtid & 31;
  const int NT = F1 / 16;                 // 8 column tiles
  int mt = wave / NT, nt = wave - mt * NT;
  if (mt >= N_NODES / 16) return;         // 3125 row tiles, exact

  int lo = lane & 15, hi = lane >> 4;
  // A fragment: lanes 0-15 row M=lo hold K {0..7, 16..23}; lanes 16-31 hold K {8..15, 24..31}
  const bf16_t* arow = A  + (size_t)(mt * 16 + lo) * IN_CH;
  // B fragment: lanes 0-15 col N=lo hold K {0..7, 16..23}; lanes 16-31 hold K {8..15, 24..31}
  const bf16_t* bcol = Bt + (size_t)(nt * 16 + lo) * IN_CH;

  f8v acc = {};
#pragma unroll
  for (int k0 = 0; k0 < IN_CH; k0 += 32) {
    bf8v a0 = *(const bf8v*)(arow + k0 + 8 * hi);
    bf8v a1 = *(const bf8v*)(arow + k0 + 16 + 8 * hi);
    bf8v b0 = *(const bf8v*)(bcol + k0 + 8 * hi);
    bf8v b1 = *(const bf8v*)(bcol + k0 + 16 + 8 * hi);
    bf16v av, bv;
#pragma unroll
    for (int i = 0; i < 8; ++i) { av[i] = a0[i]; av[i + 8] = a1[i];
                                  bv[i] = b0[i]; bv[i + 8] = b1[i]; }
    // (neg_a, A, neg_b, B, c_mod, C, reuse_a, reuse_b)
    acc = __builtin_amdgcn_wmma_f32_16x16x32_bf16(false, av, false, bv, (short)0, acc,
                                                  false, false);
  }
  // C layout: VGPR r, lanes 0-15 -> M=r, lanes 16-31 -> M=r+8; N = lane&15
  float* cp = C + (size_t)(mt * 16 + 8 * hi) * F1 + nt * 16 + lo;
#pragma unroll
  for (int r = 0; r < 8; ++r) cp[(size_t)r * F1] = acc[r];
}

// ---------------- layer-1 attention coefficients ----------------
__global__ void k_alpha1(const float* __restrict__ h, const float* __restrict__ a_s,
                         const float* __restrict__ a_d,
                         float* __restrict__ as_, float* __restrict__ ad_) {
  int i = blockIdx.x * blockDim.x + threadIdx.x;      // n*HEADS + hd
  if (i >= N_NODES * HEADS) return;
  int hd = i & (HEADS - 1);
  const float* hp = h + (size_t)i * HID;              // h is [N, HEADS, HID] flat
  float s = 0.f, d = 0.f;
#pragma unroll
  for (int c = 0; c < HID; ++c) {
    float v = hp[c];
    s += v * a_s[hd * HID + c];
    d += v * a_d[hd * HID + c];
  }
  as_[i] = s; ad_[i] = d;
}

__global__ void k_init1(unsigned* __restrict__ mx, float* __restrict__ sm,
                        float* __restrict__ acc) {
  int i = blockIdx.x * blockDim.x + threadIdx.x;
  if (i < N_NODES * HEADS) { mx[i] = NEG_INF_KEY; sm[i] = 0.f; }
  if (i < N_NODES * F1) acc[i] = 0.f;
}

// ---------------- edge passes (templated over heads / channels) ----------------
template <int H>
static __device__ __forceinline__ void edge_decode(int idx, const int* __restrict__ ei,
                                                   int& hd, int& s, int& d) {
  int e = idx / H; hd = idx - e * H;
  if (e < N_EDGES) { s = ei[e]; d = ei[N_EDGES + e]; }
  else             { s = d = e - N_EDGES; }           // self loop
}
static __device__ __forceinline__ float leaky(float v) {
  return v > 0.f ? v : NEG_SLOPE * v;
}

template <int H>
__global__ void k_edge_max(const int* __restrict__ ei, const float* __restrict__ as_,
                           const float* __restrict__ ad_, unsigned* __restrict__ mx) {
  int idx = blockIdx.x * blockDim.x + threadIdx.x;
  if (idx >= E_TOT * H) return;
  int hd, s, d; edge_decode<H>(idx, ei, hd, s, d);
  float v = leaky(as_[s * H + hd] + ad_[d * H + hd]);
  atomMaxU(&mx[d * H + hd], fkey(v));
}

template <int H>
__global__ void k_edge_sum(const int* __restrict__ ei, const float* __restrict__ as_,
                           const float* __restrict__ ad_, const unsigned* __restrict__ mx,
                           float* __restrict__ sm) {
  int idx = blockIdx.x * blockDim.x + threadIdx.x;
  if (idx >= E_TOT * H) return;
  int hd, s, d; edge_decode<H>(idx, ei, hd, s, d);
  float v = leaky(as_[s * H + hd] + ad_[d * H + hd]);
  float p = __expf(v - fkeyinv(mx[d * H + hd]));
  atomAddF(&sm[d * H + hd], p);
}

template <int H, int C>
__global__ void k_edge_aggr(const int* __restrict__ ei, const float* __restrict__ as_,
                            const float* __restrict__ ad_, const unsigned* __restrict__ mx,
                            const float* __restrict__ sm, const float* __restrict__ hfeat,
                            float* __restrict__ acc) {
  int idx = blockIdx.x * blockDim.x + threadIdx.x;
  if (idx >= E_TOT * H) return;
  int hd, s, d; edge_decode<H>(idx, ei, hd, s, d);
  float v = leaky(as_[s * H + hd] + ad_[d * H + hd]);
  float alpha = __expf(v - fkeyinv(mx[d * H + hd])) / (sm[d * H + hd] + 1e-16f);
  const float* hs = hfeat + ((size_t)s * H + hd) * C;
  float*       ap = acc   + ((size_t)d * H + hd) * C;
#pragma unroll
  for (int c = 0; c < C; ++c) atomAddF(&ap[c], alpha * hs[c]);
}

// ---------------- layer-1 finalize: h1 = elu(acc + b1) ----------------
__global__ void k_elu_bias(const float* __restrict__ acc, const float* __restrict__ b,
                           float* __restrict__ out) {
  int i = blockIdx.x * blockDim.x + threadIdx.x;
  if (i >= N_NODES * F1) return;
  float v = acc[i] + b[i & (F1 - 1)];
  out[i] = v > 0.f ? v : expm1f(v);
}

// ---------------- layer 2: [128 -> 2] projection fused with attention dots ----------------
__global__ void k_layer2_feat(const float* __restrict__ h1, const float* __restrict__ W2,
                              const float* __restrict__ a_s, const float* __restrict__ a_d,
                              float* __restrict__ h2, float* __restrict__ as_,
                              float* __restrict__ ad_) {
  int n = blockIdx.x * blockDim.x + threadIdx.x;
  if (n >= N_NODES) return;
  const float* hp = h1 + (size_t)n * F1;
  float d0 = 0.f, d1 = 0.f;
#pragma unroll 8
  for (int k = 0; k < F1; ++k) {
    float v = hp[k];
    d0 += v * W2[2 * k];
    d1 += v * W2[2 * k + 1];
  }
  h2[n * 2] = d0; h2[n * 2 + 1] = d1;
  as_[n] = d0 * a_s[0] + d1 * a_s[1];
  ad_[n] = d0 * a_d[0] + d1 * a_d[1];
}

__global__ void k_init2(unsigned* __restrict__ mx, float* __restrict__ sm,
                        float* __restrict__ acc) {
  int i = blockIdx.x * blockDim.x + threadIdx.x;
  if (i < N_NODES) { mx[i] = NEG_INF_KEY; sm[i] = 0.f; }
  if (i < N_NODES * OUT_CH) acc[i] = 0.f;
}

__global__ void k_out(const float* __restrict__ acc2, const float* __restrict__ b2,
                      float* __restrict__ out) {
  int i = blockIdx.x * blockDim.x + threadIdx.x;
  if (i >= N_NODES * OUT_CH) return;
  out[i] = acc2[i] + b2[i & 1];    // 1 head -> mean is identity
}

// ---------------- launch ----------------
static inline int cdiv(long long a, int b) { return (int)((a + b - 1) / b); }

extern "C" void kernel_launch(void* const* d_in, const int* in_sizes, int n_in,
                              void* d_out, int out_size, void* d_ws, size_t ws_size,
                              hipStream_t stream) {
  const float* x    = (const float*)d_in[0];
  const int*   ei   = (const int*)d_in[1];     // [2, E] flat: src then dst
  const float* W1   = (const float*)d_in[2];
  const float* aS1  = (const float*)d_in[3];
  const float* aD1  = (const float*)d_in[4];
  const float* b1   = (const float*)d_in[5];
  const float* W2   = (const float*)d_in[6];
  const float* aS2  = (const float*)d_in[7];
  const float* aD2  = (const float*)d_in[8];
  const float* b2   = (const float*)d_in[9];
  float* out = (float*)d_out;

  // workspace carve-up (all offsets 64B-aligned)
  char* ws = (char*)d_ws;
  size_t o = 0;
  bf16_t*   xb   = (bf16_t*)(ws + o);   o += (size_t)N_NODES * IN_CH * 2;       // 12.8 MB
  bf16_t*   w1t  = (bf16_t*)(ws + o);   o += (size_t)IN_CH * F1 * 2;            // 32 KB
  float*    h    = (float*)(ws + o);    o += (size_t)N_NODES * F1 * 4;          // 25.6 MB
  float*    acc  = (float*)(ws + o);    o += (size_t)N_NODES * F1 * 4;          // 25.6 MB
  float*    as1  = (float*)(ws + o);    o += (size_t)N_NODES * HEADS * 4;
  float*    ad1  = (float*)(ws + o);    o += (size_t)N_NODES * HEADS * 4;
  unsigned* mx1  = (unsigned*)(ws + o); o += (size_t)N_NODES * HEADS * 4;
  float*    sm1  = (float*)(ws + o);    o += (size_t)N_NODES * HEADS * 4;
  float*    h2   = (float*)(ws + o);    o += (size_t)N_NODES * OUT_CH * 4;
  float*    as2  = (float*)(ws + o);    o += (size_t)N_NODES * 4;
  float*    ad2  = (float*)(ws + o);    o += (size_t)N_NODES * 4;
  unsigned* mx2  = (unsigned*)(ws + o); o += (size_t)N_NODES * 4;
  float*    sm2  = (float*)(ws + o);    o += (size_t)N_NODES * 4;
  float*    acc2 = (float*)(ws + o);    o += (size_t)N_NODES * OUT_CH * 4;
  (void)ws_size; (void)n_in; (void)in_sizes; (void)out_size;

  const int B = 256;

  // prep: bf16 copies for the WMMA GEMM
  k_f32_to_bf16<<<cdiv((long long)N_NODES * IN_CH, B), B, 0, stream>>>(x, xb, N_NODES * IN_CH);
  k_w1_bf16_t<<<cdiv(IN_CH * F1, B), B, 0, stream>>>(W1, w1t);

  // GEMM1: 3125x8 = 25000 waves, 8 waves per 256-thread block
  k_gemm1_wmma<<<(N_NODES / 16) * (F1 / 16) / 8, B, 0, stream>>>(xb, w1t, h);

  // layer-1 attention
  k_alpha1<<<cdiv((long long)N_NODES * HEADS, B), B, 0, stream>>>(h, aS1, aD1, as1, ad1);
  k_init1<<<cdiv((long long)N_NODES * F1, B), B, 0, stream>>>(mx1, sm1, acc);

  const int EH1 = E_TOT * HEADS;
  k_edge_max<HEADS><<<cdiv(EH1, B), B, 0, stream>>>(ei, as1, ad1, mx1);
  k_edge_sum<HEADS><<<cdiv(EH1, B), B, 0, stream>>>(ei, as1, ad1, mx1, sm1);
  k_edge_aggr<HEADS, HID><<<cdiv(EH1, B), B, 0, stream>>>(ei, as1, ad1, mx1, sm1, h, acc);

  // h1 = elu(acc + b1)  (in place over h)
  k_elu_bias<<<cdiv((long long)N_NODES * F1, B), B, 0, stream>>>(acc, b1, h);

  // layer 2
  k_layer2_feat<<<cdiv(N_NODES, B), B, 0, stream>>>(h, W2, aS2, aD2, h2, as2, ad2);
  k_init2<<<cdiv((long long)N_NODES * OUT_CH, B), B, 0, stream>>>(mx2, sm2, acc2);

  k_edge_max<1><<<cdiv(E_TOT, B), B, 0, stream>>>(ei, as2, ad2, mx2);
  k_edge_sum<1><<<cdiv(E_TOT, B), B, 0, stream>>>(ei, as2, ad2, mx2, sm2);
  k_edge_aggr<1, OUT_CH><<<cdiv(E_TOT, B), B, 0, stream>>>(ei, as2, ad2, mx2, sm2, h2, acc2);

  k_out<<<cdiv((long long)N_NODES * OUT_CH, B), B, 0, stream>>>(acc2, b2, out);
}
